// ScoreDecoder_59236188946946
// MI455X (gfx1250) — matrix operand order repaired and checked by reading, not compiled, not run on hardware
//
#include <hip/hip_runtime.h>
#include <hip/hip_bf16.h>

// ---------------- types for WMMA ----------------
typedef __bf16 bf16_t;
typedef __attribute__((ext_vector_type(16))) __bf16 v16bf;
typedef __attribute__((ext_vector_type(8)))  float  v8f;

#define BATCH 64
#define CHAN  768
#define HH    24
#define WW    24
#define HWSZ  (HH*WW)   // 576
#define HID   768

// ---------------- hat CDF (piecewise quadratic) ----------------
__device__ __forceinline__ float hat_cdf(float t) {
    if (t <= -1.0f) return 0.0f;
    if (t <=  0.0f) { float u = t + 1.0f; return 0.5f * u * u; }
    if (t <=  1.0f) { float u = 1.0f - t; return 1.0f - 0.5f * u * u; }
    return 1.0f;
}

// ================= K1: per-batch telescoped bin weights + scale ==========
// Sum over pool bins telescopes: sum_q Wx[b,q,w] = hat(x1-w) - hat(x0-w).
__global__ void weights_kernel(const float* __restrict__ box,
                               float* __restrict__ scale,
                               float* __restrict__ wxs,
                               float* __restrict__ wys) {
    int b = blockIdx.x;
    int t = threadIdx.x;  // 0..63
    float x0 = box[b*4+0] * (float)WW;
    float y0 = box[b*4+1] * (float)WW;   // reference scales both by w (=24)
    float x1 = box[b*4+2] * (float)WW;
    float y1 = box[b*4+3] * (float)WW;
    if (t < WW) {
        wxs[b*WW + t] = hat_cdf(x1 - (float)t) - hat_cdf(x0 - (float)t);
    } else if (t < WW + HH) {
        int h = t - WW;
        wys[b*HH + h] = hat_cdf(y1 - (float)h) - hat_cdf(y0 - (float)h);
    } else if (t == WW + HH) {
        float bw = (x1 - x0) * 0.25f;
        float bh = (y1 - y0) * 0.25f;
        float area = bw * bh;
        scale[b] = (area > 0.0f) ? (1.0f / (16.0f * fmaxf(area, 1e-12f))) : 0.0f;
    }
}

// ================= K2: bandwidth-bound rank-1 spatial reduction ==========
// One wave per (b, c) channel; 576 contiguous floats read as float4.
// This stage streams the full 113 MB tensor and sets the kernel's runtime
// floor (~4.9 us at 23.3 TB/s).
__global__ __launch_bounds__(256)
void pool_kernel(const float* __restrict__ feat,
                 const float* __restrict__ wxs,
                 const float* __restrict__ wys,
                 const float* __restrict__ scale,
                 float* __restrict__ fvec) {
    __shared__ float sx[WW], sy[HH], comb[HWSZ];
    int b   = blockIdx.y;
    int tid = threadIdx.y * 32 + threadIdx.x;
    if (tid < WW)            sx[tid]      = wxs[b*WW + tid];
    else if (tid < WW + HH)  sy[tid - WW] = wys[b*HH + (tid - WW)];
    __syncthreads();
    for (int j = tid; j < HWSZ; j += 256) comb[j] = sy[j / WW] * sx[j % WW];
    __syncthreads();

    int c = blockIdx.x * 8 + threadIdx.y;
    const float4* p4 = (const float4*)(feat + ((size_t)(b * CHAN + c)) * HWSZ);
    float acc = 0.0f;
    int lane = threadIdx.x;
    for (int i = lane; i < HWSZ / 4; i += 32) {
        float4 v = p4[i];
        int base = i * 4;  // 4 | 24, so the quad never crosses a row
        acc += v.x * comb[base]     + v.y * comb[base + 1]
             + v.z * comb[base + 2] + v.w * comb[base + 3];
    }
    for (int off = 16; off; off >>= 1) acc += __shfl_xor(acc, off, 32);
    if (lane == 0) fvec[b * CHAN + c] = acc * scale[b];
}

// ================= K3/K4: bf16 WMMA GEMM  D = relu(A*W + bias) ===========
// A: 64x768 f32 row-major, W: 768x768 f32 row-major, out: 64x768 f32.
// All dims compile-time so B-fragment strided loads fold into immediate
// offsets (no runtime u64 mul/add chains). One wave per 16x16 tile,
// K-loop of v_wmma_f32_16x16x32_bf16, unrolled x2 for back-to-back WMMAs.
__global__ __launch_bounds__(128)
void gemm_wmma_bf16(const float* __restrict__ A,
                    const float* __restrict__ W,
                    const float* __restrict__ bias,
                    float* __restrict__ out) {
    constexpr int N = HID;   // 768
    constexpr int K = HID;   // 768

    int lane  = threadIdx.x;        // 0..31
    int mtile = threadIdx.y;        // 0..3  (M=64)
    int ntile = blockIdx.x;         // 0..47
    int hi    = lane >> 4;          // lane half
    int l15   = lane & 15;

    int arow = mtile * 16 + l15;    // A-fragment row for this lane
    int bcol = ntile * 16 + l15;    // B-fragment col for this lane

    const float* abase = A + (size_t)arow * K + 8 * hi;
    const float* wbase = W + (size_t)(8 * hi) * N + bcol;

    v8f acc = {};
    #pragma unroll 2
    for (int kb = 0; kb < K; kb += 32) {
        v16bf afrag, bfrag;
        // A 16x32 bf16 layout: lane half hi covers K {0..7,16..23}+8*hi
        const float* arp = abase + kb;
        #pragma unroll
        for (int v = 0; v < 8; ++v) {
            int k0 = ((v & 4) << 2) + 2 * (v & 3);   // 0,2,4,6,16,18,20,22
            afrag[2*v]   = (bf16_t)arp[k0];
            afrag[2*v+1] = (bf16_t)arp[k0 + 1];
        }
        // B 32x16 bf16 layout: lane half hi covers K {0..7,16..23}+8*hi
        const float* wp = wbase + (size_t)kb * N;
        #pragma unroll
        for (int v = 0; v < 8; ++v) {
            int k0 = ((v & 4) << 2) + 2 * (v & 3);
            bfrag[2*v]   = (bf16_t)wp[(k0)     * N];
            bfrag[2*v+1] = (bf16_t)wp[(k0 + 1) * N];
        }
        acc = __builtin_amdgcn_wmma_f32_16x16x32_bf16(
                  false, afrag, false, bfrag, (short)0, acc, false, false);
    }

    float bv = bias[bcol];
    #pragma unroll
    for (int r = 0; r < 8; ++r) {
        int m = mtile * 16 + 8 * hi + r;   // C/D layout: m = 8*hi + r, n = l15
        out[(size_t)m * N + bcol] = fmaxf(acc[r] + bv, 0.0f);
    }
}

// ================= K5: head  out[b] = h2[b,:] . w3 + b3 ==================
__global__ void head_kernel(const float* __restrict__ h2,
                            const float* __restrict__ w3,
                            const float* __restrict__ b3,
                            float* __restrict__ out) {
    int b = blockIdx.x;
    int lane = threadIdx.x;
    float acc = 0.0f;
    for (int c = lane; c < HID; c += 32) acc += h2[b * HID + c] * w3[c];
    for (int off = 16; off; off >>= 1) acc += __shfl_xor(acc, off, 32);
    if (lane == 0) out[b] = acc + b3[0];
}

// ================= launch ================================================
extern "C" void kernel_launch(void* const* d_in, const int* in_sizes, int n_in,
                              void* d_out, int out_size, void* d_ws, size_t ws_size,
                              hipStream_t stream) {
    const float* feat = (const float*)d_in[0];  // (64,768,24,24)
    const float* box  = (const float*)d_in[1];  // (64,4)
    const float* w1   = (const float*)d_in[2];  // (768,768)
    const float* b1   = (const float*)d_in[3];  // (768)
    const float* w2   = (const float*)d_in[4];  // (768,768)
    const float* b2   = (const float*)d_in[5];  // (768)
    const float* w3   = (const float*)d_in[6];  // (768,1)
    const float* b3   = (const float*)d_in[7];  // (1)
    float* out = (float*)d_out;                 // (64,1,1)

    float* ws    = (float*)d_ws;
    float* scale = ws;                          // 64
    float* wxs   = scale + 64;                  // 64*24
    float* wys   = wxs + BATCH * WW;            // 64*24
    float* fvec  = wys + BATCH * HH;            // 64*768
    float* h1    = fvec + BATCH * CHAN;         // 64*768
    float* h2    = h1 + BATCH * HID;            // 64*768

    // K1: bin weights (telescoped) + normalization scale
    weights_kernel<<<BATCH, 64, 0, stream>>>(box, scale, wxs, wys);

    // K2: 113 MB stream -> pooled feature vector (memory-bound stage)
    pool_kernel<<<dim3(CHAN / 8, BATCH), dim3(32, 8), 0, stream>>>(
        feat, wxs, wys, scale, fvec);

    // K3/K4: MLP layers via bf16 WMMA
    gemm_wmma_bf16<<<dim3(HID / 16), dim3(32, 4), 0, stream>>>(fvec, w1, b1, h1);
    gemm_wmma_bf16<<<dim3(HID / 16), dim3(32, 4), 0, stream>>>(h1, w2, b2, h2);

    // K5: scalar head
    head_kernel<<<BATCH, 32, 0, stream>>>(h2, w3, b3, out);
}
